// Neurcomp_39539468927581
// MI455X (gfx1250) — matrix operand order repaired and checked by reading, not compiled, not run on hardware
//
#include <hip/hip_runtime.h>
#include <hip/hip_bf16.h>

typedef __attribute__((ext_vector_type(8)))  _Float16 v8h;
typedef __attribute__((ext_vector_type(16))) _Float16 v16h;
typedef __attribute__((ext_vector_type(8)))  float    v8f;

#define OMEGA_C 30.0f
#define HID 128
#define LDH 136          // padded LDS row stride in halves (272 B, 16B-aligned)
#define MT  128          // points per workgroup

__global__ __launch_bounds__(256) void siren_wmma_kernel(
    const float* __restrict__ x,
    const float* __restrict__ W0, const float* __restrict__ b0,
    const float* __restrict__ Wa, const float* __restrict__ ba,
    const float* __restrict__ Wb, const float* __restrict__ bb,
    const float* __restrict__ Wf, const float* __restrict__ bf,
    float* __restrict__ out, int n_blocks)
{
    __shared__ __attribute__((aligned(16))) float    Hf32[MT][HID];   // fp32 residual stream
    __shared__ __attribute__((aligned(16))) _Float16 A16[MT][LDH];    // f16 matmul A input
    __shared__ __attribute__((aligned(16))) _Float16 W16[HID][LDH];   // f16 staged weights (omega folded)
    __shared__ __attribute__((aligned(16))) float    xs[MT * 3];
    __shared__ __attribute__((aligned(16))) float    Wfs[HID];

    const int tid  = threadIdx.x;
    const int lane = tid & 31;
    const int wave = tid >> 5;
    const int lh   = lane & 15;      // column / row-in-halfwave index
    const int hi   = lane >> 4;      // 0 for lanes 0-15, 1 for lanes 16-31
    const int base_pt = blockIdx.x * MT;

    // ---- stage coordinates and final-layer weights ----
    for (int i = tid; i < MT * 3; i += 256) xs[i] = x[base_pt * 3 + i];
    if (tid < HID) Wfs[tid] = Wf[tid];
    __syncthreads();

    // ---- first SineLayer: h = sin(30*(x @ W0^T + b0)), K=3 (scalar math) ----
    for (int i = tid; i < MT * HID; i += 256) {
        int m = i >> 7, n = i & (HID - 1);
        float v = fmaf(xs[m * 3 + 0], W0[n * 3 + 0],
                  fmaf(xs[m * 3 + 1], W0[n * 3 + 1],
                  fmaf(xs[m * 3 + 2], W0[n * 3 + 2], b0[n])));
        v = __sinf(OMEGA_C * v);
        Hf32[m][n] = v;
        A16[m][n]  = (_Float16)v;
    }

    const int mA = wave * 16 + lh;   // A-matrix row owned by this lane

    for (int blk = 0; blk < n_blocks; ++blk) {
        const float w1 = (blk > 0) ? 0.5f : 1.0f;
        const float w2 = (blk == n_blocks - 1) ? 0.5f : 1.0f;
        const float* bias_a = ba + blk * HID;
        const float* bias_b = bb + blk * HID;

        // ================= stage Wa: W16 = (omega*w1) * Wa =================
        {
            const float sc = OMEGA_C * w1;
            const float4* wg = (const float4*)(Wa + blk * HID * HID);
            for (int i = tid; i < HID * HID / 4; i += 256) {
                float4 w = wg[i];
                int r = (i * 4) >> 7, c = (i * 4) & (HID - 1);
                _Float16* p = &W16[r][c];
                p[0] = (_Float16)(w.x * sc); p[1] = (_Float16)(w.y * sc);
                p[2] = (_Float16)(w.z * sc); p[3] = (_Float16)(w.w * sc);
            }
        }
        __syncthreads();

        // prefetch Wb for this block into cache while matmul 1 runs
        {
            const char* nw = (const char*)(Wb + blk * HID * HID);
            #pragma unroll
            for (int i = 0; i < 2; ++i)
                __builtin_prefetch(nw + (tid + i * 256) * 128, 0, 0);
        }

        // ============ matmul 1: S1 = sin((H @ (omega*w1*Wa)^T) + omega*ba) ============
        {
            v8f acc[8] = {};
            #pragma unroll
            for (int kt = 0; kt < 4; ++kt) {
                const int ka = kt * 32 + hi * 8;
                v16h a;
                {
                    v8h lo = *(const v8h*)&A16[mA][ka];
                    v8h hh = *(const v8h*)&A16[mA][ka + 16];
                    #pragma unroll
                    for (int e = 0; e < 8; ++e) { a[e] = lo[e]; a[e + 8] = hh[e]; }
                }
                #pragma unroll
                for (int nt = 0; nt < 8; ++nt) {
                    const int nr = nt * 16 + lh;       // weight row = output neuron
                    const int kb = kt * 32 + hi * 8;   // K halves kb..kb+7 and kb+16..kb+23
                    v16h b;
                    v8h blo = *(const v8h*)&W16[nr][kb];
                    v8h bhi = *(const v8h*)&W16[nr][kb + 16];
                    #pragma unroll
                    for (int e = 0; e < 8; ++e) { b[e] = blo[e]; b[e + 8] = bhi[e]; }
                    acc[nt] = __builtin_amdgcn_wmma_f32_16x16x32_f16(
                        false, a, false, b, (short)0, acc[nt], false, false);
                }
            }
            __syncthreads();   // all waves done reading A16/W16
            #pragma unroll
            for (int nt = 0; nt < 8; ++nt) {
                int n = nt * 16 + lh;
                float bv = OMEGA_C * bias_a[n];
                #pragma unroll
                for (int v = 0; v < 8; ++v) {
                    int m = wave * 16 + v + hi * 8;    // D layout: lanes 16-31 -> M+8
                    float s = __sinf(acc[nt][v] + bv);
                    A16[m][n] = (_Float16)s;
                }
            }
        }

        // ================= stage Wb: W16 = omega * Wb =================
        {
            const float4* wg = (const float4*)(Wb + blk * HID * HID);
            for (int i = tid; i < HID * HID / 4; i += 256) {
                float4 w = wg[i];
                int r = (i * 4) >> 7, c = (i * 4) & (HID - 1);
                _Float16* p = &W16[r][c];
                p[0] = (_Float16)(w.x * OMEGA_C); p[1] = (_Float16)(w.y * OMEGA_C);
                p[2] = (_Float16)(w.z * OMEGA_C); p[3] = (_Float16)(w.w * OMEGA_C);
            }
        }
        __syncthreads();

        // prefetch next block's Wa into cache while matmul 2 runs
        if (blk + 1 < n_blocks) {
            const char* nw = (const char*)(Wa + (blk + 1) * HID * HID);
            #pragma unroll
            for (int i = 0; i < 2; ++i)
                __builtin_prefetch(nw + (tid + i * 256) * 128, 0, 0);
        }

        // ===== matmul 2 + residual: H = w2*(H + sin((S1 @ (omega*Wb)^T) + omega*bb)) =====
        {
            v8f acc[8] = {};
            #pragma unroll
            for (int kt = 0; kt < 4; ++kt) {
                const int ka = kt * 32 + hi * 8;
                v16h a;
                {
                    v8h lo = *(const v8h*)&A16[mA][ka];
                    v8h hh = *(const v8h*)&A16[mA][ka + 16];
                    #pragma unroll
                    for (int e = 0; e < 8; ++e) { a[e] = lo[e]; a[e + 8] = hh[e]; }
                }
                #pragma unroll
                for (int nt = 0; nt < 8; ++nt) {
                    const int nr = nt * 16 + lh;
                    const int kb = kt * 32 + hi * 8;
                    v16h b;
                    v8h blo = *(const v8h*)&W16[nr][kb];
                    v8h bhi = *(const v8h*)&W16[nr][kb + 16];
                    #pragma unroll
                    for (int e = 0; e < 8; ++e) { b[e] = blo[e]; b[e + 8] = bhi[e]; }
                    acc[nt] = __builtin_amdgcn_wmma_f32_16x16x32_f16(
                        false, a, false, b, (short)0, acc[nt], false, false);
                }
            }
            __syncthreads();
            #pragma unroll
            for (int nt = 0; nt < 8; ++nt) {
                int n = nt * 16 + lh;
                float bv = OMEGA_C * bias_b[n];
                #pragma unroll
                for (int v = 0; v < 8; ++v) {
                    int m = wave * 16 + v + hi * 8;
                    float s = __sinf(acc[nt][v] + bv);
                    float hn = w2 * (Hf32[m][n] + s);   // same thread owns this (m,n)
                    Hf32[m][n] = hn;
                    A16[m][n]  = (_Float16)hn;
                }
            }
        }
    }

    // ---- final linear layer: out = H @ Wf^T + bf (N=1, scalar) ----
    __syncthreads();
    if (tid < MT) {
        float accf = bf[0];
        #pragma unroll 8
        for (int n = 0; n < HID; ++n) accf = fmaf(Hf32[tid][n], Wfs[n], accf);
        out[base_pt + tid] = accf;
    }
}

extern "C" void kernel_launch(void* const* d_in, const int* in_sizes, int n_in,
                              void* d_out, int out_size, void* d_ws, size_t ws_size,
                              hipStream_t stream) {
    const float* x  = (const float*)d_in[0];
    const float* W0 = (const float*)d_in[1];
    const float* b0 = (const float*)d_in[2];
    const float* Wa = (const float*)d_in[3];
    const float* ba = (const float*)d_in[4];
    const float* Wb = (const float*)d_in[5];
    const float* bb = (const float*)d_in[6];
    const float* Wf = (const float*)d_in[7];
    const float* bf = (const float*)d_in[8];
    float* out = (float*)d_out;

    const int n_pts    = in_sizes[0] / 3;              // 262144
    const int n_blocks = in_sizes[3] / (HID * HID);    // 7
    const int grid     = n_pts / MT;                   // 2048

    siren_wmma_kernel<<<grid, 256, 0, stream>>>(x, W0, b0, Wa, ba, Wb, bb,
                                                Wf, bf, out, n_blocks);
}
